// EmformerBlock_4853313044921
// MI455X (gfx1250) — matrix operand order repaired and checked by reading, not compiled, not run on hardware
//
#include <hip/hip_runtime.h>

// ---------------------------------------------------------------------------
// Emformer block for MI455X (gfx1250, wave32, WMMA + Tensor Data Mover).
// bf16 operands + f32 accumulation on v_wmma_f32_16x16x32_bf16 for all
// GEMMs/attention; LayerNorm/softmax/residuals in f32.
// GEMMs: 64x128 workgroup tile, 4 waves, double-buffered LDS staged by TDM
// (tensor_load_to_lds + s_wait_tensorcnt), consumed via ds_load_b128.
// ---------------------------------------------------------------------------

typedef __attribute__((ext_vector_type(16))) __bf16    v16bf;
typedef __attribute__((ext_vector_type(8)))  float     v8f;
typedef __attribute__((ext_vector_type(4)))  unsigned  tdm_g0_t;   // D# group 0
typedef __attribute__((ext_vector_type(8)))  int       tdm_g1_t;   // D# group 1
typedef __attribute__((ext_vector_type(4)))  int       tdm_g23_t;  // D# groups 2/3

constexpr int Bb  = 4;
constexpr int Ss  = 2560;
constexpr int Dd  = 512;
constexpr int Hh  = 8;
constexpr int Ff  = 2048;
constexpr int DKk = 64;
constexpr int Mrows = Bb * Ss;            // 10240 token rows

constexpr int BM = 64, BN = 128, BK = 32; // GEMM workgroup tile

#define WMMA_BF16(a, b, c) \
  __builtin_amdgcn_wmma_f32_16x16x32_bf16(false, (a), false, (b), (short)0, (c), false, false)

#define HAVE_TDM __has_builtin(__builtin_amdgcn_tensor_load_to_lds)

__device__ __forceinline__ __bf16 f2bf(float f) {
  union { float f; unsigned u; } a; a.f = f;
  unsigned r = a.u + 0x7FFFu + ((a.u >> 16) & 1u);   // round-to-nearest-even
  union { unsigned short s; __bf16 b; } o; o.s = (unsigned short)(r >> 16);
  return o.b;
}

// A-fragment (16x32 bf16, ISA 7.12.2): lane holds row = lane&15.
__device__ __forceinline__ v16bf loadA(const __bf16* __restrict__ base, int ld,
                                       int r0, int k0, int lane) {
  const int r  = r0 + (lane & 15);
  const int kh = (lane >> 4) & 1;
  const __bf16* p = base + (size_t)r * ld + k0 + kh * 8;
  union { v16bf v; uint4 u[2]; } t;
  t.u[0] = *(const uint4*)p;
  t.u[1] = *(const uint4*)(p + 16);
  return t.v;
}

// B-fragment (32x16 bf16): lane holds col N; elems = K 8h..8h+7, 16+8h..16+8h+7.
// Source stored "Bt" = [N][K] row-major -> two contiguous b128 loads.
__device__ __forceinline__ v16bf loadB(const __bf16* __restrict__ base, int ld,
                                       int n0, int k0, int lane) {
  const int n  = n0 + (lane & 15);
  const int kh = (lane >> 4) & 1;
  const __bf16* p = base + (size_t)n * ld + k0 + kh * 8;
  union { v16bf v; uint4 u[2]; } t;
  t.u[0] = *(const uint4*)p;
  t.u[1] = *(const uint4*)(p + 16);
  return t.v;
}

#if HAVE_TDM
// Build a 2-D TDM descriptor (ISA ch.8: D# groups) and issue
// tensor_load_to_lds: tile = [rows] x [BK] bf16 from a row-major matrix with
// leading dimension ldElems, into contiguous LDS at ldsByte.
__device__ __forceinline__ void tdm_load_tile(const __bf16* g, unsigned ldsByte,
                                              int rows, int ldElems) {
  const unsigned long long ga = (unsigned long long)(const void*)g;
  tdm_g0_t g0 = { 1u,                                   // count=1, user mode
                  ldsByte,                              // lds_addr
                  (unsigned)ga,                         // global_addr[31:0]
                  (unsigned)(ga >> 32) | 0x80000000u }; // addr[56:32] | type=2
  tdm_g1_t g1;
  g1[0] = (int)(1u << 16);                               // data_size = 2 bytes
  g1[1] = (int)(((unsigned)ldElems & 0xFFFFu) << 16);    // tensor_dim0 lo16
  g1[2] = (int)(((unsigned)ldElems >> 16) |
                (((unsigned)rows & 0xFFFFu) << 16));     // dim0 hi16 | dim1 lo16
  g1[3] = (int)(((unsigned)rows >> 16) |
                ((unsigned)BK << 16));                   // dim1 hi16 | tile_dim0
  g1[4] = (int)(unsigned)rows;                           // tile_dim1 (tile_dim2=0)
  g1[5] = (int)(unsigned)ldElems;                        // tensor_dim0_stride lo32
  g1[6] = 0;                                             // stride0 hi16 | stride1 lo16
  g1[7] = 0;
  tdm_g23_t gz = { 0, 0, 0, 0 };
#if __clang_major__ >= 23
  tdm_g1_t gz8 = { 0, 0, 0, 0, 0, 0, 0, 0 };
  __builtin_amdgcn_tensor_load_to_lds(g0, g1, gz, gz, gz8, 0);
#else
  __builtin_amdgcn_tensor_load_to_lds(g0, g1, gz, gz, 0);
#endif
}
#endif

// ---------------------------------------------------------------------------
// Weight convert + transpose:  in [K][N] f32  ->  out [N][K] bf16
// ---------------------------------------------------------------------------
__global__ void transpose_bf16(const float* __restrict__ in, __bf16* __restrict__ out,
                               int K, int N) {
  const long long total = (long long)K * N;
  const long long idx = (long long)blockIdx.x * blockDim.x + threadIdx.x;
  if (idx >= total) return;
  const int n  = (int)(idx / K);
  const int kk = (int)(idx % K);
  out[idx] = f2bf(in[(size_t)kk * N + n]);
}

// ---------------------------------------------------------------------------
// LayerNorm over D=512: one wave per row; optional residual add; exports the
// pre-norm sum (for later residuals) and f32/bf16 normalized outputs.
// ---------------------------------------------------------------------------
__global__ void ln_kernel(const float* __restrict__ in, const float* __restrict__ addsrc,
                          const float* __restrict__ g, const float* __restrict__ be,
                          float* __restrict__ sum_out, float* __restrict__ norm_f32,
                          __bf16* __restrict__ norm_b16) {
  const int row  = blockIdx.x;
  const int lane = threadIdx.x;
  const float* p = in + (size_t)row * Dd;
  float4 v[4];
  float s = 0.f, s2 = 0.f;
#pragma unroll
  for (int i = 0; i < 4; i++) {
    const int idx = i * 128 + lane * 4;
    float4 t = *(const float4*)(p + idx);
    if (addsrc) {
      const float4 a = *(const float4*)(addsrc + (size_t)row * Dd + idx);
      t.x += a.x; t.y += a.y; t.z += a.z; t.w += a.w;
    }
    v[i] = t;
    s  += t.x + t.y + t.z + t.w;
    s2 += t.x * t.x + t.y * t.y + t.z * t.z + t.w * t.w;
  }
#pragma unroll
  for (int off = 16; off; off >>= 1) {
    s  += __shfl_xor(s,  off, 32);
    s2 += __shfl_xor(s2, off, 32);
  }
  const float mu   = s * (1.f / Dd);
  const float var  = s2 * (1.f / Dd) - mu * mu;
  const float rstd = rsqrtf(var + 1e-5f);
#pragma unroll
  for (int i = 0; i < 4; i++) {
    const int idx = i * 128 + lane * 4;
    if (sum_out) *(float4*)(sum_out + (size_t)row * Dd + idx) = v[i];
    const float4 t = v[i];
    const float o0 = (t.x - mu) * rstd * g[idx + 0] + be[idx + 0];
    const float o1 = (t.y - mu) * rstd * g[idx + 1] + be[idx + 1];
    const float o2 = (t.z - mu) * rstd * g[idx + 2] + be[idx + 2];
    const float o3 = (t.w - mu) * rstd * g[idx + 3] + be[idx + 3];
    if (norm_f32) *(float4*)(norm_f32 + (size_t)row * Dd + idx) = make_float4(o0, o1, o2, o3);
    if (norm_b16) {
      __bf16* q = norm_b16 + (size_t)row * Dd + idx;
      q[0] = f2bf(o0); q[1] = f2bf(o1); q[2] = f2bf(o2); q[3] = f2bf(o3);
    }
  }
}

// ---------------------------------------------------------------------------
// WMMA GEMM, TDM-staged: workgroup = 4 waves computes a 64x128 tile of
// C = A[M,K] x Bt[N,K]^T. Wave (wm,wn) owns a 32x64 quadrant (8 WMMAs/K-step
// from LDS). Double-buffered LDS tiles are filled by wave 0 via
// tensor_load_to_lds; consumers gate on s_wait_tensorcnt + barrier.
// MODE 0: out bf16 [M,N] (+bias)                 (FFN1)
// MODE 1: out f32  [M,N] (+bias +residual)       (FFN2)
// MODE 2: out bf16 head-split [B,H,S,64] (+bias) (Q, K)
// MODE 3: out bf16 transposed [B,H,64,S] (+bias) (V)
// ---------------------------------------------------------------------------
template <int MODE>
__global__ void gemm_wmma(const __bf16* __restrict__ A, const __bf16* __restrict__ Bt,
                          const float* __restrict__ bias, const float* __restrict__ resid,
                          void* __restrict__ outv, int M, int N, int K, int nblocks) {
  __shared__ __bf16 As[2][BM * BK] __attribute__((aligned(16)));
  __shared__ __bf16 Bs[2][BN * BK] __attribute__((aligned(16)));
  const int tid  = threadIdx.x;        // 0..127
  const int lane = tid & 31;
  const int wave = tid >> 5;           // 0..3
  const int wm = wave >> 1, wn = wave & 1;
  const int bn = blockIdx.x % nblocks;
  const int bm = blockIdx.x / nblocks;
  const int m0 = bm * BM, n0 = bn * BN;
  const __bf16* Ab = A  + (size_t)m0 * K;
  const __bf16* Bb = Bt + (size_t)n0 * K;

  auto stage = [&](int k0, int buf) {
#if HAVE_TDM
    if (wave == 0) {   // one wave drives the Tensor Data Mover (EXEC ignored)
      tdm_load_tile(Ab + k0, (unsigned)(unsigned long long)(void*)&As[buf][0], BM, K);
      tdm_load_tile(Bb + k0, (unsigned)(unsigned long long)(void*)&Bs[buf][0], BN, K);
    }
#else
#pragma unroll
    for (int c = tid; c < (BM * BK) / 8; c += 128) {
      const int r = c >> 2, off = (c & 3) * 8;
      *(uint4*)&As[buf][r * BK + off] = *(const uint4*)(Ab + (size_t)r * K + k0 + off);
    }
#pragma unroll
    for (int c = tid; c < (BN * BK) / 8; c += 128) {
      const int r = c >> 2, off = (c & 3) * 8;
      *(uint4*)&Bs[buf][r * BK + off] = *(const uint4*)(Bb + (size_t)r * K + k0 + off);
    }
#endif
  };
  auto stage_wait = [&]() {
#if HAVE_TDM
    if (wave == 0) __builtin_amdgcn_s_wait_tensorcnt(0);
#endif
    __syncthreads();
  };

  stage(0, 0);
  stage_wait();

  v8f c[2][4] = {};
  for (int k0 = 0; k0 < K; k0 += BK) {
    const int cur = (k0 / BK) & 1;
    if (k0 + BK < K) stage(k0 + BK, cur ^ 1);   // overlap TDM with WMMA
    const __bf16* Asb = &As[cur][0];
    const __bf16* Bsb = &Bs[cur][0];
    const v16bf a0 = loadA(Asb, BK, wm * 32,      0, lane);
    const v16bf a1 = loadA(Asb, BK, wm * 32 + 16, 0, lane);
#pragma unroll
    for (int nb = 0; nb < 4; nb++) {
      const v16bf b = loadB(Bsb, BK, wn * 64 + nb * 16, 0, lane);
      c[0][nb] = WMMA_BF16(a0, b, c[0][nb]);
      c[1][nb] = WMMA_BF16(a1, b, c[1][nb]);
    }
    stage_wait();
  }

  const int kh = (lane >> 4) & 1, col = lane & 15;
#pragma unroll
  for (int nb = 0; nb < 4; nb++) {
    const int n = n0 + wn * 64 + nb * 16 + col;
    const float bi = bias[n];
#pragma unroll
    for (int mb = 0; mb < 2; mb++) {
#pragma unroll
      for (int r = 0; r < 8; r++) {
        const int m = m0 + wm * 32 + mb * 16 + r + 8 * kh;  // C-layout rows
        const float val = c[mb][nb][r] + bi;
        if constexpr (MODE == 0) {
          ((__bf16*)outv)[(size_t)m * N + n] = f2bf(val);
        } else if constexpr (MODE == 1) {
          ((float*)outv)[(size_t)m * N + n] = val + resid[(size_t)m * N + n];
        } else {
          const int b = m / Ss, s = m % Ss;
          const int h = n >> 6, dk = n & 63;
          __bf16* o = (__bf16*)outv;
          if constexpr (MODE == 2)
            o[(((size_t)b * Hh + h) * Ss + s) * DKk + dk] = f2bf(val);
          else
            o[(((size_t)b * Hh + h) * DKk + dk) * Ss + s] = f2bf(val);
        }
      }
    }
  }
}

// ---------------------------------------------------------------------------
// Flash-style masked attention: one wave owns a (b,h, 16-row q tile); per
// 32-key step: 4 score WMMAs, online softmax (16-lane shuffle reductions
// matching C-layout row halves), P bounced through LDS (C->A layout),
// 4 PV WMMAs against transposed V.
// ---------------------------------------------------------------------------
__global__ void attn_wmma(const __bf16* __restrict__ q, const __bf16* __restrict__ kmat,
                          const __bf16* __restrict__ vT, const unsigned char* __restrict__ mask,
                          float* __restrict__ attn) {
  __shared__ __bf16 pbuf[16 * 32] __attribute__((aligned(16)));
  const int lane = threadIdx.x;
  const int qtiles = Ss / 16;
  const int qt = blockIdx.x % qtiles;
  const int bh = blockIdx.x / qtiles;
  const int b = bh / Hh;
  const int h = bh % Hh;
  const __bf16* qbase = q    + (size_t)bh * Ss * DKk;
  const __bf16* kbase = kmat + (size_t)bh * Ss * DKk;
  const __bf16* vbase = vT   + (size_t)bh * DKk * Ss;
  const unsigned char* mbase = mask + (size_t)b * Ss * Ss;
  const int kh = (lane >> 4) & 1, col = lane & 15;

  const v16bf qa0 = loadA(qbase, DKk, qt * 16, 0,  lane);  // dk 0..31
  const v16bf qa1 = loadA(qbase, DKk, qt * 16, 32, lane);  // dk 32..63

  float mrun[8], lrun[8];
  v8f o0 = {}, o1 = {}, o2 = {}, o3 = {};
#pragma unroll
  for (int r = 0; r < 8; r++) { mrun[r] = -1e30f; lrun[r] = 0.f; }

  for (int kt = 0; kt < Ss; kt += 32) {
    if (kt + 32 < Ss) {  // gfx1250 global_prefetch_b8 one tile ahead
      __builtin_prefetch(kbase + (size_t)(kt + 32 + (lane & 15)) * DKk, 0, 1);
      __builtin_prefetch(vbase + (size_t)(lane & 15) * Ss + kt + 32, 0, 1);
    }
    v8f s0 = {}, s1 = {};
    {
      v16bf kb = loadB(kbase, DKk, kt, 0, lane);
      s0 = WMMA_BF16(qa0, kb, s0);
      kb = loadB(kbase, DKk, kt, 32, lane);
      s0 = WMMA_BF16(qa1, kb, s0);
      kb = loadB(kbase, DKk, kt + 16, 0, lane);
      s1 = WMMA_BF16(qa0, kb, s1);
      kb = loadB(kbase, DKk, kt + 16, 32, lane);
      s1 = WMMA_BF16(qa1, kb, s1);
    }
#pragma unroll
    for (int r = 0; r < 8; r++) {
      const int qrow = qt * 16 + r + 8 * kh;
      const unsigned char mk0 = mbase[(size_t)qrow * Ss + kt + col];
      const unsigned char mk1 = mbase[(size_t)qrow * Ss + kt + 16 + col];
      float v0 = s0[r] * 0.125f; if (mk0) v0 = -1e30f;   // 1/sqrt(64)
      float v1 = s1[r] * 0.125f; if (mk1) v1 = -1e30f;
      float t = fmaxf(v0, v1);
#pragma unroll
      for (int off = 8; off; off >>= 1) t = fmaxf(t, __shfl_xor(t, off, 32));
      const float mnew  = fmaxf(mrun[r], t);
      const float alpha = __expf(mrun[r] - mnew);
      mrun[r] = mnew;
      const float e0 = __expf(v0 - mnew);
      const float e1 = __expf(v1 - mnew);
      float rs = e0 + e1;
#pragma unroll
      for (int off = 8; off; off >>= 1) rs += __shfl_xor(rs, off, 32);
      lrun[r] = lrun[r] * alpha + rs;
      o0[r] *= alpha; o1[r] *= alpha; o2[r] *= alpha; o3[r] *= alpha;
      pbuf[(r + 8 * kh) * 32 + col]      = f2bf(e0);
      pbuf[(r + 8 * kh) * 32 + 16 + col] = f2bf(e1);
    }
    __syncthreads();   // single-wave WG: orders ds stores vs b128 reads
    v16bf pa;
    {
      union { v16bf v; uint4 u[2]; } t;
      const __bf16* pp = pbuf + (lane & 15) * 32 + kh * 8;
      t.u[0] = *(const uint4*)pp;
      t.u[1] = *(const uint4*)(pp + 16);
      pa = t.v;
    }
    v16bf vb = loadB(vbase, Ss, 0, kt, lane);
    o0 = WMMA_BF16(pa, vb, o0);
    vb = loadB(vbase, Ss, 16, kt, lane);
    o1 = WMMA_BF16(pa, vb, o1);
    vb = loadB(vbase, Ss, 32, kt, lane);
    o2 = WMMA_BF16(pa, vb, o2);
    vb = loadB(vbase, Ss, 48, kt, lane);
    o3 = WMMA_BF16(pa, vb, o3);
    __syncthreads();
  }

  float invl[8];
#pragma unroll
  for (int r = 0; r < 8; r++) invl[r] = 1.f / lrun[r];
  float* obase = attn + ((size_t)b * Ss + qt * 16) * Dd + h * DKk;
#pragma unroll
  for (int r = 0; r < 8; r++) {
    float* orow = obase + (size_t)(r + 8 * kh) * Dd + col;
    orow[0]  = o0[r] * invl[r];
    orow[16] = o1[r] * invl[r];
    orow[32] = o2[r] * invl[r];
    orow[48] = o3[r] * invl[r];
  }
}

// ---------------------------------------------------------------------------
// Host orchestration
// ---------------------------------------------------------------------------
extern "C" void kernel_launch(void* const* d_in, const int* in_sizes, int n_in,
                              void* d_out, int out_size, void* d_ws, size_t ws_size,
                              hipStream_t stream) {
  (void)in_sizes; (void)n_in; (void)out_size; (void)ws_size;

  const float* x            = (const float*)d_in[0];
  const unsigned char* mask = (const unsigned char*)d_in[1];
  const float* ln_in_g = (const float*)d_in[2];
  const float* ln_in_b = (const float*)d_in[3];
  const float* Wq = (const float*)d_in[4];
  const float* bq = (const float*)d_in[5];
  const float* Wk = (const float*)d_in[6];
  const float* bk = (const float*)d_in[7];
  const float* Wv = (const float*)d_in[8];
  const float* bv = (const float*)d_in[9];
  const float* ln1_g = (const float*)d_in[10];
  const float* ln1_b = (const float*)d_in[11];
  const float* W1 = (const float*)d_in[12];
  const float* b1 = (const float*)d_in[13];
  const float* W2 = (const float*)d_in[14];
  const float* b2 = (const float*)d_in[15];
  const float* ln2_g = (const float*)d_in[16];
  const float* ln2_b = (const float*)d_in[17];
  float* out = (float*)d_out;

  size_t off = 0;
  auto alloc = [&](size_t bytes) -> char* {
    char* p = (char*)d_ws + off;
    off += (bytes + 255) & ~(size_t)255;
    return p;
  };
  float*  xn_f   = (float*)alloc((size_t)Mrows * Dd * 4);
  __bf16* xn_b   = (__bf16*)alloc((size_t)Mrows * Dd * 2);
  __bf16* qb     = (__bf16*)alloc((size_t)Mrows * Dd * 2);
  __bf16* kb     = (__bf16*)alloc((size_t)Mrows * Dd * 2);
  __bf16* vTb    = (__bf16*)alloc((size_t)Mrows * Dd * 2);
  float*  attn_f = (float*)alloc((size_t)Mrows * Dd * 4);
  float*  attno  = (float*)alloc((size_t)Mrows * Dd * 4);
  __bf16* hn_b   = (__bf16*)alloc((size_t)Mrows * Dd * 2);
  __bf16* h1_b   = (__bf16*)alloc((size_t)Mrows * Ff * 2);
  float*  h2r    = (float*)alloc((size_t)Mrows * Dd * 4);
  __bf16* Wtq    = (__bf16*)alloc((size_t)Dd * Dd * 2);
  __bf16* Wtk    = (__bf16*)alloc((size_t)Dd * Dd * 2);
  __bf16* Wtv    = (__bf16*)alloc((size_t)Dd * Dd * 2);
  __bf16* Wt1    = (__bf16*)alloc((size_t)Dd * Ff * 2);
  __bf16* Wt2    = (__bf16*)alloc((size_t)Ff * Dd * 2);

  // Weights: f32 [K][N] -> bf16 [N][K] (once per launch; tiny vs GEMMs).
  transpose_bf16<<<(Dd * Dd + 255) / 256, 256, 0, stream>>>(Wq, Wtq, Dd, Dd);
  transpose_bf16<<<(Dd * Dd + 255) / 256, 256, 0, stream>>>(Wk, Wtk, Dd, Dd);
  transpose_bf16<<<(Dd * Dd + 255) / 256, 256, 0, stream>>>(Wv, Wtv, Dd, Dd);
  transpose_bf16<<<(Dd * Ff + 255) / 256, 256, 0, stream>>>(W1, Wt1, Dd, Ff);
  transpose_bf16<<<(Ff * Dd + 255) / 256, 256, 0, stream>>>(W2, Wt2, Ff, Dd);

  // 1) input LayerNorm -> xn (f32 + bf16)
  ln_kernel<<<Mrows, 32, 0, stream>>>(x, nullptr, ln_in_g, ln_in_b,
                                      nullptr, xn_f, xn_b);

  // 2) QKV projections (head-split Q/K, transposed V)
  const int gQKV = (Mrows / BM) * (Dd / BN);
  gemm_wmma<2><<<gQKV, 128, 0, stream>>>(xn_b, Wtq, bq, nullptr, qb,  Mrows, Dd, Dd, Dd / BN);
  gemm_wmma<2><<<gQKV, 128, 0, stream>>>(xn_b, Wtk, bk, nullptr, kb,  Mrows, Dd, Dd, Dd / BN);
  gemm_wmma<3><<<gQKV, 128, 0, stream>>>(xn_b, Wtv, bv, nullptr, vTb, Mrows, Dd, Dd, Dd / BN);

  // 3) masked flash attention
  attn_wmma<<<Bb * Hh * (Ss / 16), 32, 0, stream>>>(qb, kb, vTb, mask, attn_f);

  // 4) residual (attn + xn) kept in f32, LN1 -> bf16
  ln_kernel<<<Mrows, 32, 0, stream>>>(attn_f, xn_f, ln1_g, ln1_b,
                                      attno, nullptr, hn_b);

  // 5) FFN
  gemm_wmma<0><<<(Mrows / BM) * (Ff / BN), 128, 0, stream>>>(hn_b, Wt1, b1, nullptr, h1_b, Mrows, Ff, Dd, Ff / BN);
  gemm_wmma<1><<<(Mrows / BM) * (Dd / BN), 128, 0, stream>>>(h1_b, Wt2, b2, attno,  h2r,  Mrows, Dd, Ff, Dd / BN);

  // 6) LN2 -> output (f32)
  ln_kernel<<<Mrows, 32, 0, stream>>>(h2r, nullptr, ln2_g, ln2_b,
                                      nullptr, out, nullptr);
}
